// Mukara_29326036697606
// MI455X (gfx1250) — matrix operand order repaired and checked by reading, not compiled, not run on hardware
//
#include <hip/hip_runtime.h>
#include <cstdint>

#define O_N 128
#define D_N 128
#define FLSOA 64
#define T_N 128
#define FE 16
#define H_N 64
#define NH 128
#define NO 64
#define PH 128
#define HS 72   // padded LDS row stride (bf16 elems): 144B, 16B-aligned offsets

typedef __bf16 bf16;
typedef bf16  bf16x16 __attribute__((ext_vector_type(16)));
typedef bf16  bf16x8  __attribute__((ext_vector_type(8)));
typedef float f32x8   __attribute__((ext_vector_type(8)));

__device__ __forceinline__ f32x8 wmma_bf16(bf16x16 a, bf16x16 b, f32x8 c) {
  // D = A(16x32 bf16) * B(32x16 bf16) + C(16x16 f32)
  return __builtin_amdgcn_wmma_f32_16x16x32_bf16(false, a, false, b, (short)0, c,
                                                 false, false);
}

__device__ __forceinline__ f32x8 zero8() {
  f32x8 z = {0.f, 0.f, 0.f, 0.f, 0.f, 0.f, 0.f, 0.f};
  return z;
}

#define LOG2E 1.4426950408889634f

// sigmoid via hw exp2 + hw rcp (no IEEE division expansion)
__device__ __forceinline__ float fast_sigmoid(float x) {
  return __builtin_amdgcn_rcpf(1.f + __builtin_amdgcn_exp2f(-LOG2E * x));
}

// tanh via hw v_tanh_f32 when available, else exp2+rcp identity
__device__ __forceinline__ float fast_tanh(float x) {
#if __has_builtin(__builtin_amdgcn_tanhf)
  return __builtin_amdgcn_tanhf(x);
#else
  // tanh(x) = 1 - 2/(exp(2x)+1)
  return 1.f - 2.f * __builtin_amdgcn_rcpf(1.f + __builtin_amdgcn_exp2f(2.f * LOG2E * x));
#endif
}

// Build a B-fragment (32x16 bf16) from row-major W[K][ldn], col tile at nbase,
// K chunk at kbase; rows >= kmax are zero-padded.
// Layout: lane l holds N = nbase + (l&15); VGPR v holds K = kbase + (l>=16?8:0) + 2v + (v>=4?8:0), +1.
__device__ __forceinline__ bf16x16 make_bfrag(const float* __restrict__ W, int ldn,
                                              int kbase, int nbase, int kmax, int lane) {
  int n = nbase + (lane & 15);
  int koff = (lane >> 4) * 8;
  bf16x16 f;
#pragma unroll
  for (int v = 0; v < 8; ++v) {
    int k0 = kbase + koff + 2 * v + (v >= 4 ? 8 : 0);
    float a = (k0 < kmax)     ? W[(size_t)k0 * ldn + n]       : 0.f;
    float b = (k0 + 1 < kmax) ? W[(size_t)(k0 + 1) * ldn + n] : 0.f;
    f[2 * v]     = (bf16)a;
    f[2 * v + 1] = (bf16)b;
  }
  return f;
}

// ---------------------------------------------------------------------------
// Kernel 1: node encoders (tiny; VALU). blocks 0..127 -> eo rows, 128..255 -> ed.
// ---------------------------------------------------------------------------
__global__ void enc_kernel(const float* __restrict__ Of, const float* __restrict__ Df,
                           const float* __restrict__ WO1, const float* __restrict__ bO1,
                           const float* __restrict__ WO2, const float* __restrict__ bO2,
                           const float* __restrict__ WO3, const float* __restrict__ bO3,
                           const float* __restrict__ WD1, const float* __restrict__ bD1,
                           const float* __restrict__ WD2, const float* __restrict__ bD2,
                           const float* __restrict__ WD3, const float* __restrict__ bD3,
                           float* __restrict__ eo, float* __restrict__ ed) {
  __shared__ float xs[FLSOA];
  __shared__ float h1[NH];
  __shared__ float h2[NH];
  int b = blockIdx.x;
  bool isO = b < O_N;
  int row = isO ? b : b - O_N;
  const float* x  = isO ? Of : Df;
  const float* W1 = isO ? WO1 : WD1; const float* b1 = isO ? bO1 : bD1;
  const float* W2 = isO ? WO2 : WD2; const float* b2 = isO ? bO2 : bD2;
  const float* W3 = isO ? WO3 : WD3; const float* b3 = isO ? bO3 : bD3;
  float* out = isO ? eo : ed;
  int tid = threadIdx.x;

  if (tid < FLSOA) xs[tid] = x[row * FLSOA + tid];
  __syncthreads();
  {
    float acc = b1[tid];
    for (int k = 0; k < FLSOA; ++k) acc = fmaf(xs[k], W1[k * NH + tid], acc);
    h1[tid] = fmaxf(acc, 0.f);
  }
  __syncthreads();
  {
    float acc = b2[tid];
    for (int k = 0; k < NH; ++k) acc = fmaf(h1[k], W2[k * NH + tid], acc);
    h2[tid] = fmaxf(acc, 0.f);
  }
  __syncthreads();
  if (tid < NO) {
    float acc = b3[tid];
    for (int k = 0; k < NH; ++k) acc = fmaf(h2[k], W3[k * NO + tid], acc);
    out[row * NO + tid] = acc;
  }
}

// ---------------------------------------------------------------------------
// Kernel 2: Apair = eo @ Wp1[:64,:], BT[n][j] = (ed @ Wp1[64:,:])[j][n]
// ---------------------------------------------------------------------------
__global__ void pairpart_kernel(const float* __restrict__ eo, const float* __restrict__ ed,
                                const float* __restrict__ Wp1,
                                float* __restrict__ Apair, float* __restrict__ BT) {
  __shared__ float v[NO];
  int b = blockIdx.x, tid = threadIdx.x;
  bool isA = b < O_N;
  int row = isA ? b : b - O_N;
  const float* src = isA ? eo : ed;
  if (tid < NO) v[tid] = src[row * NO + tid];
  __syncthreads();
  const float* W = Wp1 + (isA ? 0 : (size_t)NO * PH);
  float acc = 0.f;
  for (int k = 0; k < NO; ++k) acc = fmaf(v[k], W[k * PH + tid], acc);
  if (isA) Apair[row * PH + tid] = acc;
  else     BT[tid * D_N + row]  = acc;  // transposed for coalesced reads later
}

// ---------------------------------------------------------------------------
// Kernel 3: s_pair[i][j] = softplus( sum_n relu(A[i,n]+B[j,n]+bp1[n]) * Wp2[n] + bp2 )
// ---------------------------------------------------------------------------
__global__ void spair_kernel(const float* __restrict__ Apair, const float* __restrict__ BT,
                             const float* __restrict__ bp1, const float* __restrict__ Wp2,
                             const float* __restrict__ bp2, float* __restrict__ s_pair) {
  __shared__ float Ash[PH], bsh[PH], wsh[PH];
  int i = blockIdx.x, j = threadIdx.x;
  Ash[j] = Apair[i * PH + j];
  bsh[j] = bp1[j];
  wsh[j] = Wp2[j];
  __syncthreads();
  float acc = bp2[0];
  for (int n = 0; n < PH; ++n)
    acc = fmaf(fmaxf(Ash[n] + BT[n * D_N + j] + bsh[n], 0.f), wsh[n], acc);
  float sp = (acc > 20.f) ? acc : log1pf(__expf(acc));
  s_pair[i * D_N + j] = sp;
}

// ---------------------------------------------------------------------------
// Kernel 4: length-masked GRU over 16384 sequences, WMMA bf16, 16 rows / wave.
// Per step: gates[16,192] = x_t@W_ih + h@W_hh via 36 v_wmma_f32_16x16x32_bf16.
// Weights live in registers (gfx1250 VGPR-MSB space); h round-trips through a
// wave-private LDS tile to be reshaped from C-layout to A-fragment layout.
// x stream is double-buffered in registers with global_prefetch look-ahead.
// Epilogue fuses sigmoid(h@Ws+bs)*s_pair and atomically reduces into out[0].
// ---------------------------------------------------------------------------
__global__ __launch_bounds__(128, 1) void gru_kernel(
    const float* __restrict__ seqs, const int* __restrict__ lengths,
    const float* __restrict__ W_ih, const float* __restrict__ b_ih,
    const float* __restrict__ W_hh, const float* __restrict__ b_hh,
    const float* __restrict__ Ws, const float* __restrict__ bs,
    const float* __restrict__ s_pair, float* __restrict__ out) {
  __shared__ alignas(16) bf16 hlds[4][16][HS];

  int lane = threadIdx.x & 31;
  int wave = threadIdx.x >> 5;
  int p0 = (blockIdx.x * 4 + wave) * 16;
  int nn = lane & 15;
  int half = lane >> 4;
  int koff = half * 8;

  // Register-resident weight B-fragments. Column tiles: [0..3]=r, [4..7]=z, [8..11]=n.
  bf16x16 Wih[12], Wh0[12], Wh1[12];
#pragma unroll
  for (int c = 0; c < 12; ++c) {
    Wih[c] = make_bfrag(W_ih, 192, 0,  c * 16, FE,  lane);  // K=16 padded to 32
    Wh0[c] = make_bfrag(W_hh, 192, 0,  c * 16, H_N, lane);  // K chunk 0..31
    Wh1[c] = make_bfrag(W_hh, 192, 32, c * 16, H_N, lane);  // K chunk 32..63
  }

  // Per-lane gate biases (depend only on N = nn within each 16-col chunk).
  float brz_r[4], brz_z[4], bxn[4], bhn[4];
#pragma unroll
  for (int c = 0; c < 4; ++c) {
    brz_r[c] = b_ih[c * 16 + nn]       + b_hh[c * 16 + nn];
    brz_z[c] = b_ih[64 + c * 16 + nn]  + b_hh[64 + c * 16 + nn];
    bxn[c]   = b_ih[128 + c * 16 + nn];
    bhn[c]   = b_hh[128 + c * 16 + nn];
  }

  // Sequence lengths for the 8 accumulator rows owned by this lane half.
  int len_e[8];
#pragma unroll
  for (int e = 0; e < 8; ++e) len_e[e] = lengths[p0 + half * 8 + e];

  // h in f32 C-layout: h[c] covers cols [c*16, c*16+16), rows via (e, half).
  f32x8 h[4];
#pragma unroll
  for (int c = 0; c < 4; ++c)
#pragma unroll
    for (int e = 0; e < 8; ++e) h[c][e] = 0.f;

  // Zero the wave-private LDS h tile (h0 = 0).
#pragma unroll
  for (int c = 0; c < 4; ++c)
#pragma unroll
    for (int e = 0; e < 8; ++e)
      hlds[wave][half * 8 + e][c * 16 + nn] = (bf16)0.f;

  const float* xrow = seqs + ((size_t)(p0 + nn) * T_N) * FE + koff;

  // x double-buffer: issue loads for step t+1 before computing step t.
  float4 xa = ((const float4*)xrow)[0];
  float4 xb = ((const float4*)xrow)[1];

  for (int t = 0; t < T_N; ++t) {
    float4 na = xa, nb = xb;
    if (t + 1 < T_N) {
      const float4* xp = (const float4*)(xrow + (size_t)(t + 1) * FE);
      na = xp[0];
      nb = xp[1];
    }
    if (t + 8 < T_N)
      __builtin_prefetch(xrow + (size_t)(t + 8) * FE, 0, 1);  // global_prefetch_b8

    // x_t A-fragment: 8 contiguous f32 per lane, K padded 16->32 with zeros.
    bf16x16 xf;
    xf[0] = (bf16)xa.x; xf[1] = (bf16)xa.y; xf[2] = (bf16)xa.z; xf[3] = (bf16)xa.w;
    xf[4] = (bf16)xb.x; xf[5] = (bf16)xb.y; xf[6] = (bf16)xb.z; xf[7] = (bf16)xb.w;
#pragma unroll
    for (int e = 8; e < 16; ++e) xf[e] = (bf16)0.f;

    // h A-fragments (two K chunks of 32) from LDS, 16B vector loads.
    bf16x16 hA0, hA1;
    {
      bf16x8 lo0 = *(const bf16x8*)&hlds[wave][nn][koff];
      bf16x8 hi0 = *(const bf16x8*)&hlds[wave][nn][16 + koff];
      bf16x8 lo1 = *(const bf16x8*)&hlds[wave][nn][32 + koff];
      bf16x8 hi1 = *(const bf16x8*)&hlds[wave][nn][48 + koff];
#pragma unroll
      for (int e = 0; e < 8; ++e) {
        hA0[e] = lo0[e]; hA0[8 + e] = hi0[e];
        hA1[e] = lo1[e]; hA1[8 + e] = hi1[e];
      }
    }

#pragma unroll
    for (int c = 0; c < 4; ++c) {
      f32x8 ar = zero8();
      ar = wmma_bf16(xf,  Wih[c], ar);
      ar = wmma_bf16(hA0, Wh0[c], ar);
      ar = wmma_bf16(hA1, Wh1[c], ar);
      f32x8 az = zero8();
      az = wmma_bf16(xf,  Wih[4 + c], az);
      az = wmma_bf16(hA0, Wh0[4 + c], az);
      az = wmma_bf16(hA1, Wh1[4 + c], az);
      f32x8 axn = zero8();
      axn = wmma_bf16(xf,  Wih[8 + c], axn);
      f32x8 ahn = zero8();
      ahn = wmma_bf16(hA0, Wh0[8 + c], ahn);
      ahn = wmma_bf16(hA1, Wh1[8 + c], ahn);

#pragma unroll
      for (int e = 0; e < 8; ++e) {
        float r = fast_sigmoid(ar[e] + brz_r[c]);
        float z = fast_sigmoid(az[e] + brz_z[c]);
        float n = fast_tanh(axn[e] + bxn[c] + r * (ahn[e] + bhn[c]));
        float hold = h[c][e];
        float hnew = (1.f - z) * n + z * hold;
        float hv = (t < len_e[e]) ? hnew : hold;
        h[c][e] = hv;
        hlds[wave][half * 8 + e][c * 16 + nn] = (bf16)hv;
      }
    }

    xa = na;
    xb = nb;
  }

  // Epilogue: p_seq = sigmoid(h_last @ Ws + bs), contribution = p_seq * s_pair.
  float contrib = 0.f;
  if (lane < 16) {
    float acc = bs[0];
#pragma unroll
    for (int k = 0; k < H_N; ++k)
      acc = fmaf((float)hlds[wave][lane][k], Ws[k], acc);
    float pseq = fast_sigmoid(acc);
    contrib = pseq * s_pair[p0 + lane];
  }
#pragma unroll
  for (int off = 16; off > 0; off >>= 1)
    contrib += __shfl_down(contrib, off, 32);
  if (lane == 0) atomicAdd(out, contrib);
}

// ---------------------------------------------------------------------------
extern "C" void kernel_launch(void* const* d_in, const int* in_sizes, int n_in,
                              void* d_out, int out_size, void* d_ws, size_t ws_size,
                              hipStream_t stream) {
  (void)in_sizes; (void)n_in; (void)out_size; (void)ws_size;
  const float* Of   = (const float*)d_in[0];
  const float* Df   = (const float*)d_in[1];
  const float* seqs = (const float*)d_in[2];
  const int*   lens = (const int*)d_in[3];
  const float* WO1 = (const float*)d_in[4];  const float* bO1 = (const float*)d_in[5];
  const float* WO2 = (const float*)d_in[6];  const float* bO2 = (const float*)d_in[7];
  const float* WO3 = (const float*)d_in[8];  const float* bO3 = (const float*)d_in[9];
  const float* WD1 = (const float*)d_in[10]; const float* bD1 = (const float*)d_in[11];
  const float* WD2 = (const float*)d_in[12]; const float* bD2 = (const float*)d_in[13];
  const float* WD3 = (const float*)d_in[14]; const float* bD3 = (const float*)d_in[15];
  const float* Wp1 = (const float*)d_in[16]; const float* bp1 = (const float*)d_in[17];
  const float* Wp2 = (const float*)d_in[18]; const float* bp2 = (const float*)d_in[19];
  const float* Wih = (const float*)d_in[20]; const float* bih = (const float*)d_in[21];
  const float* Whh = (const float*)d_in[22]; const float* bhh = (const float*)d_in[23];
  const float* Wsq = (const float*)d_in[24]; const float* bsq = (const float*)d_in[25];

  float* ws     = (float*)d_ws;
  float* eo     = ws;                       // 128*64
  float* ed     = eo + O_N * NO;            // 128*64
  float* Apair  = ed + D_N * NO;            // 128*128
  float* BT     = Apair + O_N * PH;         // 128*128 (transposed)
  float* spair  = BT + PH * D_N;            // 128*128

  enc_kernel<<<256, 128, 0, stream>>>(Of, Df, WO1, bO1, WO2, bO2, WO3, bO3,
                                      WD1, bD1, WD2, bD2, WD3, bD3, eo, ed);
  pairpart_kernel<<<256, 128, 0, stream>>>(eo, ed, Wp1, Apair, BT);
  spair_kernel<<<128, 128, 0, stream>>>(Apair, BT, bp1, Wp2, bp2, spair);
  hipMemsetAsync(d_out, 0, sizeof(float), stream);
  gru_kernel<<<256, 128, 0, stream>>>(seqs, lens, Wih, bih, Whh, bhh,
                                      Wsq, bsq, spair, (float*)d_out);
}
